// RaftSe3_85109071937858
// MI455X (gfx1250) — matrix operand order repaired and checked by reading, not compiled, not run on hardware
//
#include <hip/hip_runtime.h>

// ---------------------------------------------------------------------------
// MI455X (gfx1250) RAFT-SE3 update block.
// ~650 GFLOP conv work vs ~0.3 GB I/O -> compute bound: everything runs as
// bf16 implicit GEMM on V_WMMA_F32_16X16X32_BF16 with f32 accumulation.
//  * Weight (A) tiles DMA'd into LDS by the Tensor Data Mover
//    (tensor_load_to_lds, 2-D descriptor, OOB-zero = free tail padding),
//    issued by wave 0 and overlapped with im2col B staging by all waves,
//    synchronized with s_wait_tensorcnt + workgroup barrier.
//  * LDS B tile stored in *fragment order* so every lane loads its 16 bf16
//    fragment elements as two ds_load_b128; A fragments are two b128 loads
//    + shufflevector (native 16-bit A layout is two contiguous 16B runs).
// ---------------------------------------------------------------------------

typedef __bf16 v16bf __attribute__((ext_vector_type(16)));
typedef __bf16 v8bf  __attribute__((ext_vector_type(8)));
typedef float  v8f   __attribute__((ext_vector_type(8)));
typedef unsigned int u32x4 __attribute__((ext_vector_type(4)));
typedef int          i32x4 __attribute__((ext_vector_type(4)));
typedef int          i32x8 __attribute__((ext_vector_type(8)));

#define HW   64
#define PIXN 4096          // 64*64 pixels per image
#define BN_IMGS 32         // B*N
#define BS_IMGS 16         // B*S

// ------------------------------ conversion kernels -------------------------

__global__ void cvt_flat(const float* __restrict__ s, __bf16* __restrict__ d, long n) {
  long i = (long)blockIdx.x * blockDim.x + threadIdx.x;
  if (i < n) d[i] = (__bf16)s[i];
}

__global__ void cvt_into(const float* __restrict__ s, __bf16* __restrict__ d,
                         int C, int dstC, int chOff, long n) {
  long i = (long)blockIdx.x * blockDim.x + threadIdx.x;
  if (i < n) {
    long img = i / ((long)C * PIXN);
    long rem = i - img * (long)C * PIXN;
    d[(img * (long)dstC + chOff) * PIXN + rem] = (__bf16)s[i];
  }
}

// ------------------------------ WMMA conv kernel ---------------------------
// D[Cout x 4096] = W[Cout x Cin*KS*KS] * im2col(src).  Block tile 64x128,
// 8 waves; per 32-deep k-chunk each wave reuses one A fragment across four
// B fragments (4 WMMAs).
// ST: 0 bf16 [img,dstC,pix]+chOff | 1 f32 [img,Cout,pix]
//     2 f32 [(img*4096+pix)*3+m]  | 3 f32 0.01*softplus at [img*4096+pix]
// ACT: 0 none, 1 relu, 2 sigmoid, 3 tanh
template<int KS, int PAD, int ACT, int ST>
__global__ __launch_bounds__(256)
void conv_wmma(const __bf16* __restrict__ src, int Cin,
               const __bf16* __restrict__ Wb, const float* __restrict__ bias,
               int Cout,
               __bf16* __restrict__ dstb, float* __restrict__ dstf,
               int dstC, int dstOff) {
  const int Ktot = Cin * KS * KS;

  // A tile: row-major 64 x 32 bf16 (TDM destination, stride 32, OOB zeroed)
  __shared__ alignas(32) __bf16 lA[64 * 32];
  // B tile: fragment order: [8 subtiles][32 lanes][16 elems]
  __shared__ alignas(32) __bf16 lB[8 * 32 * 16];

  const int tid  = threadIdx.x;
  const int img  = blockIdx.z;
  const int n0   = blockIdx.x * 128;   // pixel tile base
  const int m0   = blockIdx.y * 64;    // cout tile base

  const size_t srcBase = (size_t)img * Cin * PIXN;

  const int lane = tid & 31, wid = tid >> 5;
  const int wm = wid & 3;              // 4 waves across M (16 rows each)
  const int wn = wid >> 2;             // 2 waves across N (64 px each)
  const int half = lane >> 4, l15 = lane & 15;

  const unsigned ldsA = (unsigned)(size_t)(void*)lA;  // LDS byte offset (addr[31:0])

  v8f acc[4] = {};

  for (int k0 = 0; k0 < Ktot; k0 += 32) {
    // ---- wave 0: TDM async copy of the 64x32 weight tile into LDS --------
    if (wid == 0) {
      size_t ga = (size_t)(const void*)(Wb + (size_t)m0 * Ktot + k0);
      unsigned td0 = (unsigned)(Ktot - k0);   // remaining k (OOB -> zero fill)
      unsigned td1 = (unsigned)(Cout - m0);   // remaining rows
      u32x4 g0;
      g0[0] = 1u;                                        // count=1, user mode
      g0[1] = ldsA;                                      // lds_addr
      g0[2] = (unsigned)ga;                              // global_addr[31:0]
      g0[3] = ((unsigned)(ga >> 32) & 0x01FFFFFFu) | (2u << 30); // [56:32], type=2
      i32x8 g1;
      g1[0] = 0x00010000;                                // data_size=2B
      g1[1] = (int)((td0 & 0xFFFFu) << 16);              // tensor_dim0 lo
      g1[2] = (int)(((td0 >> 16) & 0xFFFFu) | ((td1 & 0xFFFFu) << 16));
      g1[3] = (int)(((td1 >> 16) & 0xFFFFu) | (32u << 16)); // tile_dim0 = 32
      g1[4] = 64;                                        // tile_dim1=64, tile_dim2=0
      g1[5] = Ktot;                                      // tensor_dim0_stride lo32
      g1[6] = 0;                                         // stride hi, dim1_stride lo
      g1[7] = 0;
      i32x4 gz4 = {0, 0, 0, 0};
      i32x8 gz8 = {0, 0, 0, 0, 0, 0, 0, 0};
      __builtin_amdgcn_tensor_load_to_lds(g0, g1, gz4, gz4, gz8, 0);
    }

    // ---- all waves: im2col gather of B tile, stored in fragment order ----
    for (int t = tid; t < 32 * 128; t += 256) {
      int kr = t >> 7, pc = t & 127;
      int k = k0 + kr;
      __bf16 v = (__bf16)0.0f;
      if (k < Ktot) {
        int cin = k / (KS * KS);
        int rem = k - cin * (KS * KS);
        int ky = rem / KS, kx = rem - ky * KS;
        int pix = n0 + pc;
        int y = (pix >> 6) + ky - PAD;
        int x = (pix & 63) + kx - PAD;
        if ((unsigned)y < (unsigned)HW && (unsigned)x < (unsigned)HW)
          v = src[srcBase + (size_t)cin * PIXN + y * HW + x];
      }
      // fragment-order scatter: lane=((kr>>3)&1)*16+(pc&15), sub=pc>>4, j=(kr&7)|((kr>>4)<<3)
      int sLane = (((kr >> 3) & 1) << 4) | (pc & 15);
      int sSub  = pc >> 4;
      int sJ    = (kr & 7) | ((kr >> 4) << 3);
      lB[((sSub * 32 + sLane) << 4) + sJ] = v;
    }

    if (wid == 0) __builtin_amdgcn_s_wait_tensorcnt(0);
    __syncthreads();

    // ---- A fragment: two contiguous 16B runs per lane -> 2x ds_load_b128
    {
      const __bf16* ap = lA + ((wm * 16 + l15) * 32 + half * 8);
      v8bf lo = *(const v8bf*)ap;          // kl = 8h .. 8h+7   (j = 0..7)
      v8bf hi = *(const v8bf*)(ap + 16);   // kl = 16+8h ..+7   (j = 8..15)
      v16bf a = __builtin_shufflevector(lo, hi, 0, 1, 2, 3, 4, 5, 6, 7,
                                        8, 9, 10, 11, 12, 13, 14, 15);
      // ---- four B fragments (contiguous 32B per lane) / 4 WMMAs ----------
#pragma unroll
      for (int s = 0; s < 4; ++s) {
        v16bf b = *(const v16bf*)(lB + (((wn * 4 + s) * 32 + lane) << 4));
        acc[s] = __builtin_amdgcn_wmma_f32_16x16x32_bf16(
            false, a, false, b, (short)0, acc[s], false, false);
      }
    }
    __syncthreads();
  }

  // ---- fused epilogue: bias + activation + store
#pragma unroll
  for (int s = 0; s < 4; ++s) {
#pragma unroll
    for (int r8 = 0; r8 < 8; ++r8) {
      int m = m0 + wm * 16 + r8 + 8 * half;   // C/D layout: M = vgpr + 8*half
      int pix = n0 + wn * 64 + s * 16 + l15;  // N = lane&15
      if (m < Cout) {
        float v = acc[s][r8] + bias[m];
        if constexpr (ACT == 1) v = fmaxf(v, 0.0f);
        else if constexpr (ACT == 2) v = 1.0f / (1.0f + __expf(-v));
        else if constexpr (ACT == 3) v = tanhf(v);
        if constexpr (ST == 0)
          dstb[((size_t)img * dstC + dstOff + m) * PIXN + pix] = (__bf16)v;
        else if constexpr (ST == 1)
          dstf[((size_t)img * dstC + m) * PIXN + pix] = v;
        else if constexpr (ST == 2)
          dstf[((size_t)img * PIXN + pix) * 3 + m] = v;
        else if constexpr (ST == 3)
          dstf[(size_t)img * PIXN + pix] =
              0.01f * (fmaxf(v, 0.0f) + log1pf(__expf(-fabsf(v))));
      }
    }
  }
}

// ------------------------------ elementwise kernels ------------------------

__global__ void gru_fuse(const float* __restrict__ z, const float* __restrict__ q,
                         const float* __restrict__ x,
                         float* __restrict__ h, __bf16* __restrict__ hb, long n) {
  long i = (long)blockIdx.x * blockDim.x + threadIdx.x;
  if (i < n) {
    float zv = z[i];
    float hv = (1.0f - zv) * x[i] + zv * q[i];
    h[i] = hv;
    hb[i] = (__bf16)hv;
  }
}

__global__ void build_xi2(const float* __restrict__ r, const float* __restrict__ x,
                          const __bf16* __restrict__ xi, __bf16* __restrict__ xi2,
                          long n) {
  long i = (long)blockIdx.x * blockDim.x + threadIdx.x;
  if (i < n) {
    long img = i / (512L * PIXN);
    long rem = i - img * 512L * PIXN;
    if (rem < 128L * PIXN) {
      long j = img * 128L * PIXN + rem;
      xi2[i] = (__bf16)(r[j] * x[j]);
    } else {
      xi2[i] = xi[i];
    }
  }
}

__global__ void seg_mean(const __bf16* __restrict__ a, const int* __restrict__ ii,
                         __bf16* __restrict__ m, long n) {
  long i = (long)blockIdx.x * blockDim.x + threadIdx.x;
  if (i < n) {
    long b   = i / (8L * 128 * PIXN);
    long rem = i - b * (8L * 128 * PIXN);
    int  s   = (int)(rem / (128L * PIXN));
    long cp  = rem - (long)s * 128 * PIXN;   // c*PIXN + px
    float acc = 0.0f; int cnt = 0;
    for (int nn = 0; nn < 16; ++nn) {
      if (ii[nn] == s) { acc += (float)a[(b * 16 + nn) * 128L * PIXN + cp]; ++cnt; }
    }
    m[i] = (__bf16)(cnt ? acc / (float)cnt : 0.0f);
  }
}

// ------------------------------ host orchestration -------------------------

extern "C" void kernel_launch(void* const* d_in, const int* in_sizes, int n_in,
                              void* d_out, int out_size, void* d_ws, size_t ws_size,
                              hipStream_t stream) {
  (void)in_sizes; (void)n_in; (void)out_size; (void)ws_size;

  const float* net   = (const float*)d_in[0];
  const float* inp   = (const float*)d_in[1];
  const float* corr  = (const float*)d_in[2];
  const float* resid = (const float*)d_in[3];
  const float* dzp   = (const float*)d_in[4];
  const int*   ii    = (const int*)d_in[5];
  auto Wp = [&](int j) { return (const float*)d_in[6 + 2 * j]; };
  auto Bp = [&](int j) { return (const float*)d_in[7 + 2 * j]; };
  // j: 0 corr1, 1 corr2, 2 flow1, 3 flow2, 4 dz1, 5 dz2,
  //    6 gruz, 7 grur, 8 gruq, 9 delta1, 10 delta2, 11 wt1, 12 wt2,
  //    13 agg1, 14 agg2, 15 eta
  static const int Co[16] = {128,128,128, 64,128, 64,128,128,128,128,  3,128,  3,128,128,  1};
  static const int Ci[16] = {198,128,  3,128,  1,128,512,512,512,128,128,128,128,128,128,128};
  static const int Kk[16] = {  1,  3,  7,  3,  7,  3,  3,  3,  3,  3,  3,  3,  3,  3,  3,  3};

  char* ws = (char*)d_ws;
  size_t off = 0;
  auto alloc = [&](size_t bytes) {
    size_t o = off; off += (bytes + 255) & ~(size_t)255; return o;
  };

  __bf16* wb[16];
  long wn_[16];
  for (int j = 0; j < 16; ++j) {
    wn_[j] = (long)Co[j] * Ci[j] * Kk[j] * Kk[j];
    wb[j] = (__bf16*)(ws + alloc((size_t)wn_[j] * 2));
  }
  __bf16* corr_b  = (__bf16*)(ws + alloc((size_t)BN_IMGS * 198 * PIXN * 2));
  __bf16* resid_b = (__bf16*)(ws + alloc((size_t)BN_IMGS * 3   * PIXN * 2));
  __bf16* dz_b    = (__bf16*)(ws + alloc((size_t)BN_IMGS * 1   * PIXN * 2));
  __bf16* xi      = (__bf16*)(ws + alloc((size_t)BN_IMGS * 512 * PIXN * 2));
  __bf16* xi2     = (__bf16*)(ws + alloc((size_t)BN_IMGS * 512 * PIXN * 2));
  float*  zf      = (float*) (ws + alloc((size_t)BN_IMGS * 128 * PIXN * 4));
  float*  rf      = (float*) (ws + alloc((size_t)BN_IMGS * 128 * PIXN * 4)); // reused as q
  __bf16* tmp128  = (__bf16*)(ws + alloc((size_t)BN_IMGS * 128 * PIXN * 2)); // c1/f1/d1/t/a/m
  __bf16* h_b     = (__bf16*)(ws + alloc((size_t)BN_IMGS * 128 * PIXN * 2));
  __bf16* m_in    = (__bf16*)(ws + alloc((size_t)BS_IMGS * 128 * PIXN * 2));

  float* out_h     = (float*)d_out;                 // [2,16,128,64,64]
  float* out_delta = out_h + 16777216;              // [2,16,64,64,3]
  float* out_wt    = out_delta + 393216;            // [2,16,64,64,3]
  float* out_eta   = out_wt + 393216;               // [2,8,64,64]

  auto g1 = [](long n) { return dim3((unsigned)((n + 255) / 256)); };
  auto cg = [](int Cout, int nimg) { return dim3(32, (unsigned)((Cout + 63) / 64), (unsigned)nimg); };

  // ---- 1) one-time bf16 conversion ----
  for (int j = 0; j < 16; ++j)
    cvt_flat<<<g1(wn_[j]), 256, 0, stream>>>(Wp(j), wb[j], wn_[j]);
  cvt_flat<<<g1((long)BN_IMGS * 198 * PIXN), 256, 0, stream>>>(corr,  corr_b,  (long)BN_IMGS * 198 * PIXN);
  cvt_flat<<<g1((long)BN_IMGS * 3   * PIXN), 256, 0, stream>>>(resid, resid_b, (long)BN_IMGS * 3   * PIXN);
  cvt_flat<<<g1((long)BN_IMGS * 1   * PIXN), 256, 0, stream>>>(dzp,   dz_b,    (long)BN_IMGS * 1   * PIXN);
  cvt_into<<<g1((long)BN_IMGS * 128 * PIXN), 256, 0, stream>>>(net, xi, 128, 512, 0,   (long)BN_IMGS * 128 * PIXN);
  cvt_into<<<g1((long)BN_IMGS * 128 * PIXN), 256, 0, stream>>>(inp, xi, 128, 512, 128, (long)BN_IMGS * 128 * PIXN);

  // ---- 2) encoders -> concat buffer xi = [x, inp, c, f, d] ----
  conv_wmma<1,0,1,0><<<cg(128, BN_IMGS), 256, 0, stream>>>(corr_b, 198, wb[0], Bp(0), 128, tmp128, nullptr, 128, 0);
  conv_wmma<3,1,1,0><<<cg(128, BN_IMGS), 256, 0, stream>>>(tmp128, 128, wb[1], Bp(1), 128, xi, nullptr, 512, 256);
  conv_wmma<7,3,1,0><<<cg(128, BN_IMGS), 256, 0, stream>>>(resid_b,  3, wb[2], Bp(2), 128, tmp128, nullptr, 128, 0);
  conv_wmma<3,1,1,0><<<cg( 64, BN_IMGS), 256, 0, stream>>>(tmp128, 128, wb[3], Bp(3),  64, xi, nullptr, 512, 384);
  conv_wmma<7,3,1,0><<<cg(128, BN_IMGS), 256, 0, stream>>>(dz_b,     1, wb[4], Bp(4), 128, tmp128, nullptr, 128, 0);
  conv_wmma<3,1,1,0><<<cg( 64, BN_IMGS), 256, 0, stream>>>(tmp128, 128, wb[5], Bp(5),  64, xi, nullptr, 512, 448);

  // ---- 3) ConvGRU ----
  conv_wmma<3,1,2,1><<<cg(128, BN_IMGS), 256, 0, stream>>>(xi, 512, wb[6], Bp(6), 128, nullptr, zf, 128, 0);
  conv_wmma<3,1,2,1><<<cg(128, BN_IMGS), 256, 0, stream>>>(xi, 512, wb[7], Bp(7), 128, nullptr, rf, 128, 0);
  build_xi2<<<g1((long)BN_IMGS * 512 * PIXN), 256, 0, stream>>>(rf, net, xi, xi2, (long)BN_IMGS * 512 * PIXN);
  conv_wmma<3,1,3,1><<<cg(128, BN_IMGS), 256, 0, stream>>>(xi2, 512, wb[8], Bp(8), 128, nullptr, rf, 128, 0); // q reuses rf
  gru_fuse<<<g1((long)BN_IMGS * 128 * PIXN), 256, 0, stream>>>(zf, rf, net, out_h, h_b, (long)BN_IMGS * 128 * PIXN);

  // ---- 4) delta / weight heads (transposed fp32 stores) ----
  conv_wmma<3,1,1,0><<<cg(128, BN_IMGS), 256, 0, stream>>>(h_b, 128, wb[ 9], Bp( 9), 128, tmp128, nullptr, 128, 0);
  conv_wmma<3,1,0,2><<<cg(  3, BN_IMGS), 256, 0, stream>>>(tmp128, 128, wb[10], Bp(10), 3, nullptr, out_delta, 0, 0);
  conv_wmma<3,1,1,0><<<cg(128, BN_IMGS), 256, 0, stream>>>(h_b, 128, wb[11], Bp(11), 128, tmp128, nullptr, 128, 0);
  conv_wmma<3,1,0,2><<<cg(  3, BN_IMGS), 256, 0, stream>>>(tmp128, 128, wb[12], Bp(12), 3, nullptr, out_wt, 0, 0);

  // ---- 5) GraphAgg: agg1 -> segment mean -> agg2 -> eta ----
  conv_wmma<3,1,1,0><<<cg(128, BN_IMGS), 256, 0, stream>>>(h_b, 128, wb[13], Bp(13), 128, tmp128, nullptr, 128, 0);
  seg_mean<<<g1((long)2 * 8 * 128 * PIXN), 256, 0, stream>>>(tmp128, ii, m_in, (long)2 * 8 * 128 * PIXN);
  conv_wmma<3,1,1,0><<<cg(128, BS_IMGS), 256, 0, stream>>>(m_in, 128, wb[14], Bp(14), 128, tmp128, nullptr, 128, 0);
  conv_wmma<3,1,0,3><<<cg(  1, BS_IMGS), 256, 0, stream>>>(tmp128, 128, wb[15], Bp(15), 1, nullptr, out_eta, 0, 0);
}
